// BalOpt_91302414778872
// MI455X (gfx1250) — matrix operand order repaired and checked by reading, not compiled, not run on hardware
//
#include <hip/hip_runtime.h>
#include <hip/hip_bf16.h>
#include <math.h>

#define N_NODES 100000
#define DIM 256
#define HEADS 8
#define PROTOS 64
#define HD 32
#define K2 8
#define K1 4096
#define ATTN_COLS 512   // H*P
#define W_COLS 768      // ATTN_COLS + DIM (fused attn|xv weight)

typedef __attribute__((ext_vector_type(16))) __bf16 v16bf;
typedef __attribute__((ext_vector_type(8)))  __bf16 v8bf;
typedef __attribute__((ext_vector_type(8)))  float  v8f;

__device__ inline float sigmoidf_(float x){ return 1.0f/(1.0f+__expf(-x)); }

__device__ inline void split_bf16(float f, __bf16 &hi, __bf16 &lo){
  hi = (__bf16)f;
  lo = (__bf16)(f - (float)hi);
}
__device__ inline unsigned pack2(__bf16 a, __bf16 b){
  union { __bf16 h[2]; unsigned u; } u; u.h[0]=a; u.h[1]=b; return u.u;
}
// monotonic sortable key for f32
__device__ inline unsigned fkey(float f){
  unsigned u = __float_as_uint(f);
  return (u & 0x80000000u) ? ~u : (u | 0x80000000u);
}
// build a v16bf fragment from two 16-byte LDS chunks
__device__ inline v16bf frag_from2(const __bf16* p0, const __bf16* p1){
  v8bf a = *(const v8bf*)p0;
  v8bf b = *(const v8bf*)p1;
  v16bf r;
#pragma unroll
  for (int i=0;i<8;i++){ r[i]=a[i]; r[i+8]=b[i]; }
  return r;
}

// ---------------- Kernel 1: fused weight prep ----------------
// Wfused[d][col] : col<512 -> (Wq . key_p)/sqrt(HD) ; col>=512 -> Wv
__global__ void prep_weights(const float* __restrict__ Wq, const float* __restrict__ bq,
                             const float* __restrict__ key_p,
                             const float* __restrict__ Wv, const float* __restrict__ bv,
                             __bf16* __restrict__ Whi, __bf16* __restrict__ Wlo,
                             float* __restrict__ bqk)
{
  int col = blockIdx.x;   // 0..767
  int d   = threadIdx.x;  // 0..255
  float w, b = 0.f;
  if (col < ATTN_COLS) {
    int h = col >> 6, r = col & 63;
    const float scale = 0.17677669529663687f; // 1/sqrt(32)
    float acc = 0.f, bacc = 0.f;
#pragma unroll
    for (int e=0;e<HD;e++){
      float kp = key_p[(r*HEADS + h)*HD + e];
      acc  += Wq[d*DIM + h*HD + e] * kp;
      bacc += bq[h*HD + e] * kp;
    }
    w = acc*scale; b = bacc*scale;
  } else {
    int c = col - ATTN_COLS;
    w = Wv[d*DIM + c];
    b = bv[c];
  }
  __bf16 hi, lo; split_bf16(w, hi, lo);
  Whi[d*W_COLS + col] = hi;
  Wlo[d*W_COLS + col] = lo;
  if (d==0) bqk[col] = b;
}

// ---------------- Kernel 2: fused GEMM  x(100k x 256) @ W(256 x 768) -----------
// grid: (col-block FAST, row-block) so the 12 blocks sharing one x row-panel run
// back-to-back and hit L2 (x is read from HBM ~once instead of 12x).
// cols 0..511 -> attn written TRANSPOSED as attn_t[(h*64+r)][n]
//   + fused per-(node,head) top-K2 / masked softmax (BN==64==PROTOS, so an attn
//     col-block holds the full prototype row for head h=bn for 64 nodes)
// cols 512..767 -> xv written row-major xv[n][c]
#define BM 64
#define BN 64
#define BK 32
#define LDA 40  // padded bf16 row stride (80B, 16B-aligned)

__global__ __launch_bounds__(256)
void gemm_fused(const float* __restrict__ x,
                const __bf16* __restrict__ Whi, const __bf16* __restrict__ Wlo,
                const float* __restrict__ bqk, const float* __restrict__ bias,
                float* __restrict__ attn_t, float* __restrict__ xv,
                float* __restrict__ p8, int* __restrict__ idx2)
{
  __shared__ __bf16 Ah[2][BM][LDA];
  __shared__ __bf16 Al[2][BM][LDA];
  __shared__ __bf16 Bh[2][BN][LDA];
  __shared__ __bf16 Bl[2][BN][LDA];
  __shared__ float  Cst[BN][BM+1];

  const int t    = threadIdx.x;
  const int lane = t & 31;
  const int w    = t >> 5;   // 0..7
  const int wm   = w >> 1;   // 0..3 (row tile)
  const int wn   = w & 1;    // 0..1 (col tile pair)
  const int bn   = blockIdx.x;          // col-block 0..11 (fast dim -> L2 reuse)
  const int row0 = blockIdx.y * BM;
  const int col0 = bn * BN;

  v8f acc0 = {}; v8f acc1 = {};

  auto load_stage = [&](int kt, int buf) {
    const int k0 = kt * BK;
    // A: 64 rows x 32 k, fp32 -> hi/lo bf16 split, packed as k-pairs
#pragma unroll
    for (int i = 0; i < 4; i++) {
      int pe = i*256 + t;
      int r = pe >> 4, kp = pe & 15;
      int gr = row0 + r;
      float2 v = (gr < N_NODES) ? *(const float2*)(x + (size_t)gr*DIM + k0 + 2*kp)
                                : make_float2(0.f, 0.f);
      __bf16 h0,l0,h1,l1; split_bf16(v.x,h0,l0); split_bf16(v.y,h1,l1);
      *(unsigned*)&Ah[buf][r][2*kp] = pack2(h0,h1);
      *(unsigned*)&Al[buf][r][2*kp] = pack2(l0,l1);
    }
    // B: 32 k x 64 n, store transposed in LDS as [n][k]
#pragma unroll
    for (int i = 0; i < 4; i++) {
      int pe = i*256 + t;
      int nn = pe & 63, kp = pe >> 6;
      size_t g0 = (size_t)(k0 + 2*kp)*W_COLS + col0 + nn;
      *(unsigned*)&Bh[buf][nn][2*kp] = pack2(Whi[g0], Whi[g0 + W_COLS]);
      *(unsigned*)&Bl[buf][nn][2*kp] = pack2(Wlo[g0], Wlo[g0 + W_COLS]);
    }
  };

  load_stage(0, 0);
  __syncthreads();
#pragma unroll
  for (int kt = 0; kt < DIM/BK; kt++) {
    int buf = kt & 1;
    if (kt + 1 < DIM/BK) load_stage(kt+1, buf^1);
    // fragments
    const int ar   = wm*16 + (lane & 15);
    const int kOff = (lane & 16) ? 8 : 0;     // A half selector
    v16bf a_h = frag_from2(&Ah[buf][ar][kOff], &Ah[buf][ar][kOff+16]);
    v16bf a_l = frag_from2(&Al[buf][ar][kOff], &Al[buf][ar][kOff+16]);
    const int kb  = (lane & 16) ? 8 : 0;      // B half selector
    const int bc0 = wn*32 + (lane & 15);
    const int bc1 = bc0 + 16;
    v16bf b0h = frag_from2(&Bh[buf][bc0][kb], &Bh[buf][bc0][kb+16]);
    v16bf b0l = frag_from2(&Bl[buf][bc0][kb], &Bl[buf][bc0][kb+16]);
    v16bf b1h = frag_from2(&Bh[buf][bc1][kb], &Bh[buf][bc1][kb+16]);
    v16bf b1l = frag_from2(&Bl[buf][bc1][kb], &Bl[buf][bc1][kb+16]);
    // 3-product bf16 emulation of fp32 GEMM (hi*lo + lo*hi + hi*hi)
    acc0 = __builtin_amdgcn_wmma_f32_16x16x32_bf16(false, a_h, false, b0l, (short)0, acc0, false, false);
    acc0 = __builtin_amdgcn_wmma_f32_16x16x32_bf16(false, a_l, false, b0h, (short)0, acc0, false, false);
    acc0 = __builtin_amdgcn_wmma_f32_16x16x32_bf16(false, a_h, false, b0h, (short)0, acc0, false, false);
    acc1 = __builtin_amdgcn_wmma_f32_16x16x32_bf16(false, a_h, false, b1l, (short)0, acc1, false, false);
    acc1 = __builtin_amdgcn_wmma_f32_16x16x32_bf16(false, a_l, false, b1h, (short)0, acc1, false, false);
    acc1 = __builtin_amdgcn_wmma_f32_16x16x32_bf16(false, a_h, false, b1h, (short)0, acc1, false, false);
    __syncthreads();
  }

  // add fused bias
  float bb0 = bqk[col0 + wn*32 + (lane & 15)];
  float bb1 = bqk[col0 + wn*32 + 16 + (lane & 15)];
#pragma unroll
  for (int j=0;j<8;j++){ acc0[j] += bb0; acc1[j] += bb1; }

  if (bn >= 8) {
    // xv region: direct row-major coalesced store
    int cbase = (bn-8)*64 + wn*32 + (lane & 15);
#pragma unroll
    for (int j=0;j<8;j++){
      int m = row0 + wm*16 + j + ((lane & 16) ? 8 : 0);
      if (m < N_NODES) {
        xv[(size_t)m*DIM + cbase]      = acc0[j];
        xv[(size_t)m*DIM + cbase + 16] = acc1[j];
      }
    }
  } else {
    // attn region: transpose through LDS, store column(hr)-major,
    // and fuse the per-(node, head=bn) top-K2 + masked softmax.
    int cl0 = wn*32 + (lane & 15);
#pragma unroll
    for (int j=0;j<8;j++){
      int ml = wm*16 + j + ((lane & 16) ? 8 : 0);
      Cst[cl0][ml]      = acc0[j];
      Cst[cl0+16][ml]   = acc1[j];
    }
    __syncthreads();
    // 1) coalesced transposed store of raw attn
    {
      int colL = t >> 2, seg = t & 3;
      int gcol = col0 + colL;
#pragma unroll
      for (int i=0;i<16;i++){
        int m = row0 + seg*16 + i;
        if (m < N_NODES) attn_t[(size_t)gcol*N_NODES + m] = Cst[colL][seg*16+i];
      }
    }
    // 2) top-K2 over the 64 prototypes of head h=bn for the 64 nodes in LDS
    if (t < BM) {
      int n = row0 + t;
      if (n < N_NODES) {
        int h = bn;
        float sc[K2], av[K2]; int id[K2];
#pragma unroll
        for (int k=0;k<K2;k++){ sc[k] = -1e30f; av[k] = 0.f; id[k] = 0; }
        for (int r=0;r<PROTOS;r++){
          float a = Cst[r][t];                    // bank-conflict-free: (r+t)%64
          float s = sigmoidf_(a) + bias[h*PROTOS + r];
          if (s > sc[K2-1]) {                     // strict > keeps lower index on ties
            int j = K2-1;
            while (j > 0 && s > sc[j-1]) { sc[j]=sc[j-1]; av[j]=av[j-1]; id[j]=id[j-1]; j--; }
            sc[j]=s; av[j]=a; id[j]=r;
          }
        }
        float m = av[0];
#pragma unroll
        for (int k=1;k<K2;k++) m = fmaxf(m, av[k]);
        float sum=0.f, e[K2];
#pragma unroll
        for (int k=0;k<K2;k++){ e[k]=__expf(av[k]-m); sum+=e[k]; }
        float inv = 1.0f/sum;
#pragma unroll
        for (int k=0;k<K2;k++){
          p8  [((size_t)n*HEADS + h)*K2 + k] = e[k]*inv;
          idx2[((size_t)n*HEADS + h)*K2 + k] = id[k];
        }
      }
    }
  }
}

// ---------------- Kernel 3: top-K1 threshold per (h,r) via 2-pass histogram ----
__global__ void select_threshold(const float* __restrict__ attn_t,
                                 unsigned* __restrict__ thrKey)
{
  __shared__ unsigned hist[1024];
  __shared__ int sh_bin, sh_rank;
  int hr = blockIdx.x;
  const float* colp = attn_t + (size_t)hr * N_NODES;
  int t = threadIdx.x;
  for (int i=t;i<1024;i+=256) hist[i]=0;
  __syncthreads();
  for (int n=t;n<N_NODES;n+=256)
    atomicAdd(&hist[fkey(colp[n])>>22], 1u);
  __syncthreads();
  if (t==0){
    unsigned cum=0; int b=1023;
    for (; b>0; b--){ if (cum + hist[b] >= K1) break; cum += hist[b]; }
    sh_bin = b; sh_rank = K1 - (int)cum;
  }
  __syncthreads();
  int bsel = sh_bin, rank = sh_rank;
  for (int i=t;i<1024;i+=256) hist[i]=0;
  __syncthreads();
  for (int n=t;n<N_NODES;n+=256){
    unsigned key = fkey(colp[n]);
    if ((int)(key>>22)==bsel) atomicAdd(&hist[(key>>12)&1023], 1u);
  }
  __syncthreads();
  if (t==0){
    unsigned cum=0; int s=1023;
    for (; s>0; s--){ if (cum + hist[s] >= (unsigned)rank) break; cum += hist[s]; }
    thrKey[hr] = ((unsigned)bsel<<22) | ((unsigned)s<<12);
  }
}

// ---------------- Kernel 4: prototype aggregation v[r,h,d] via WMMA -----------
#define EB 512  // nodes per block

__global__ __launch_bounds__(256)
void proto_agg(const float* __restrict__ attn_t, const float* __restrict__ xv,
               const unsigned* __restrict__ thrKey, const float* __restrict__ emb,
               float* __restrict__ vproto)
{
  __shared__ __bf16 Sl[PROTOS][LDA];  // s tile [r][n32]
  __shared__ __bf16 Xl[HD][LDA];      // xv tile transposed [d][n32]
  int h = blockIdx.y;
  int nbase0 = blockIdx.x * EB;
  int t = threadIdx.x, lane = t & 31, w = t >> 5;
  int rt = w >> 1, dt = w & 1;        // 8 waves -> 4x2 grid of 16x16 tiles
  v8f acc = {};
  for (int ks = 0; ks < EB/32; ks++) {
    int nb = nbase0 + ks*32;
    __syncthreads();
    // s = mask2 ? sigmoid(attn+emb) : 0   (mask2 = key >= threshold key)
#pragma unroll
    for (int i=0;i<4;i++){
      int pe = i*256 + t;
      int r = pe >> 4, np = pe & 15;
      int n = nb + 2*np;
      unsigned tk = thrKey[h*PROTOS + r];
      float eb_ = emb[h*PROTOS + r];
      float s0 = 0.f, s1 = 0.f;
      if (n < N_NODES) {
        float a0 = attn_t[(size_t)(h*PROTOS+r)*N_NODES + n];
        if (fkey(a0) >= tk) s0 = sigmoidf_(a0 + eb_);
      }
      if (n+1 < N_NODES) {
        float a1 = attn_t[(size_t)(h*PROTOS+r)*N_NODES + n+1];
        if (fkey(a1) >= tk) s1 = sigmoidf_(a1 + eb_);
      }
      *(unsigned*)&Sl[r][2*np] = pack2((__bf16)s0, (__bf16)s1);
    }
    // xv tile 32n x 32d -> Xl[d][n]
#pragma unroll
    for (int i=0;i<4;i++){
      int e = i*256 + t;
      int nn = e >> 5, d = e & 31;
      int n = nb + nn;
      float xvv = (n < N_NODES) ? xv[(size_t)n*DIM + h*HD + d] : 0.f;
      Xl[d][nn] = (__bf16)xvv;
    }
    __syncthreads();
    int ar = rt*16 + (lane & 15);
    int kOff = (lane & 16) ? 8 : 0;
    v16bf a = frag_from2(&Sl[ar][kOff], &Sl[ar][kOff+16]);
    int bc = dt*16 + (lane & 15);
    int kb = (lane & 16) ? 8 : 0;
    v16bf b = frag_from2(&Xl[bc][kb], &Xl[bc][kb+16]);
    acc = __builtin_amdgcn_wmma_f32_16x16x32_bf16(false, a, false, b, (short)0, acc, false, false);
  }
  int d = dt*16 + (lane & 15);
#pragma unroll
  for (int j=0;j<8;j++){
    int r = rt*16 + j + ((lane & 16) ? 8 : 0);
    atomicAdd(&vproto[((size_t)r*HEADS + h)*HD + d], acc[j]);
  }
}

// ---------------- Kernel 5: gather + finalize ----------------
#define FNODES 32

__global__ __launch_bounds__(256)
void finalize_k(const float* __restrict__ xv, const float* __restrict__ p8,
                const int* __restrict__ idx2, const float* __restrict__ vproto,
                const float* __restrict__ alpha, const float* __restrict__ beta,
                float* __restrict__ out)
{
  int t = threadIdx.x;
  int h = t >> 5, d = t & 31;
  float sa = sigmoidf_(alpha[h]);
  float sb = sigmoidf_(beta[h]);
  int n0 = blockIdx.x * FNODES;
  for (int i=0;i<FNODES;i++){
    int n = n0 + i;
    if (n >= N_NODES) break;
    const float* pp = p8   + ((size_t)n*HEADS + h)*K2;
    const int*   ii = idx2 + ((size_t)n*HEADS + h)*K2;
    float acc = 0.f;
#pragma unroll
    for (int k=0;k<K2;k++)
      acc += pp[k] * vproto[((size_t)ii[k]*HEADS + h)*HD + d];  // v table is 64KB, L2-hot
    out[(size_t)n*DIM + t] = sa * xv[(size_t)n*DIM + t] + sb * acc;
  }
}

// ---------------- launch ----------------
extern "C" void kernel_launch(void* const* d_in, const int* in_sizes, int n_in,
                              void* d_out, int out_size, void* d_ws, size_t ws_size,
                              hipStream_t stream)
{
  const float* x     = (const float*)d_in[0];
  const float* bias  = (const float*)d_in[1];
  const float* Wq    = (const float*)d_in[2];
  const float* bq    = (const float*)d_in[3];
  const float* key_p = (const float*)d_in[4];
  const float* Wv    = (const float*)d_in[5];
  const float* bv    = (const float*)d_in[6];
  const float* emb   = (const float*)d_in[7];
  const float* alpha = (const float*)d_in[8];
  const float* beta  = (const float*)d_in[9];

  float* out  = (float*)d_out;
  int*   idx2 = (int*)d_out + (size_t)N_NODES*DIM;   // second tuple element (int32)

  char* ws = (char*)d_ws;
  size_t off = 0;
  auto alloc = [&](size_t bytes){ void* p = ws + off; off += (bytes + 255) & ~(size_t)255; return p; };
  __bf16*   Whi    = (__bf16*)  alloc((size_t)DIM*W_COLS*2);
  __bf16*   Wlo    = (__bf16*)  alloc((size_t)DIM*W_COLS*2);
  float*    bqk    = (float*)   alloc(W_COLS*4);
  float*    attn_t = (float*)   alloc((size_t)ATTN_COLS*N_NODES*4);   // 204.8 MB
  float*    xv     = (float*)   alloc((size_t)N_NODES*DIM*4);         // 102.4 MB
  float*    p8     = (float*)   alloc((size_t)N_NODES*HEADS*K2*4);    //  25.6 MB
  unsigned* thr    = (unsigned*)alloc(ATTN_COLS*4);
  float*    vproto = (float*)   alloc((size_t)PROTOS*HEADS*HD*4);

  prep_weights<<<W_COLS, 256, 0, stream>>>(Wq, bq, key_p, Wv, bv, Whi, Wlo, bqk);

  // col-block fast-varying -> x row panel reused out of L2 by 12 consecutive blocks
  dim3 gg(W_COLS/BN, (N_NODES + BM - 1)/BM);
  gemm_fused<<<gg, 256, 0, stream>>>(x, Whi, Wlo, bqk, bias, attn_t, xv, p8, idx2);

  select_threshold<<<ATTN_COLS, 256, 0, stream>>>(attn_t, thr);

  hipMemsetAsync(vproto, 0, (size_t)PROTOS*HEADS*HD*4, stream);
  dim3 ge((N_NODES + EB - 1)/EB, HEADS);
  proto_agg<<<ge, 256, 0, stream>>>(attn_t, xv, thr, emb, vproto);

  finalize_k<<<(N_NODES + FNODES - 1)/FNODES, 256, 0, stream>>>(
      xv, p8, idx2, vproto, alpha, beta, out);
}
